// CartTensorMix2_85744727097802
// MI455X (gfx1250) — matrix operand order repaired and checked by reading, not compiled, not run on hardware
//
#include <hip/hip_runtime.h>
#include <hip/hip_bf16.h>
#include <math.h>

typedef _Float16 f16;
typedef __attribute__((ext_vector_type(16))) _Float16 v16h;
typedef __attribute__((ext_vector_type(8)))  float    v8f;

#define R2I 0.70710678118654752f   /* 1/sqrt(2) */
#define R6I 0.40824829046386302f   /* 1/sqrt(6) */

// ---------------------------------------------------------------------------
// WMMA wrapper: D = A(16x32 f16) * B(32x16 f16) + C(f32)
// ---------------------------------------------------------------------------
__device__ inline v8f wmma_f16(v16h a, v16h b, v8f c) {
  return __builtin_amdgcn_wmma_f32_16x16x32_f16(
      /*neg_a=*/false, a, /*neg_b=*/false, b,
      /*c_mod=*/(short)0, c, /*reuse_a=*/false, /*reuse_b=*/false);
}

__device__ inline v8f zero8() {
  v8f z = {0.f,0.f,0.f,0.f,0.f,0.f,0.f,0.f};
  return z;
}

// A-fragment (16x32, f16) gathered from f32 memory with element stride.
// `row` points at this lane's row, element k of chunk at row[k*stride].
// ISA layout: half=0 -> K = {0..7, 16..23}; half=1 -> K = {8..15, 24..31}
__device__ inline v16h load_a_strided(const float* row, int half, int stride) {
  v16h a;
#pragma unroll
  for (int e = 0; e < 8; ++e) {
    a[e]     = (f16)row[(8*half + e) * stride];
    a[8 + e] = (f16)row[(16 + 8*half + e) * stride];
  }
  return a;
}

// A-fragment from per-wave LDS f16 row (contiguous K).
__device__ inline v16h load_a_lds(const f16* row, int half, int koff) {
  v16h a;
#pragma unroll
  for (int e = 0; e < 8; ++e) {
    a[e]     = row[koff + 8*half + e];
    a[8 + e] = row[koff + 16 + 8*half + e];
  }
  return a;
}

// B-fragment: column n = lane%16, elements over K in table order (same as A).
// `r` = chunk base + lane*ld + n-offset; ld = row stride of the weight matrix.
__device__ inline v16h load_brow(const float* r, int ld) {
  v16h b;
  const int l = threadIdx.x & 31, h = l >> 4, n = l & 15;
  const float* base = r - (size_t)l * ld;
#pragma unroll
  for (int e = 0; e < 8; ++e) {
    b[e]     = (f16)base[(8*h + e) * ld + n];
    b[8 + e] = (f16)base[(16 + 8*h + e) * ld + n];
  }
  return b;
}

__device__ __host__ inline float epsf(int i, int j, int k) {
  return 0.5f * (float)((i - j) * (j - k) * (k - i));
}

// ---------------------------------------------------------------------------
// Setup: zero the (G,9) accumulator; build w3j(2,2,1) [5x5x3] and
// w3j(2,2,2) [5x5x5] from the orthonormal symmetric-traceless basis S_m.
// Basis index order is (y,z,x) to match the reference's CART=[2,0,1].
// ---------------------------------------------------------------------------
__global__ void setup_kernel(float* acc, int accN, float* w221, float* w222) {
  int idx = blockIdx.x * blockDim.x + threadIdx.x;
  int stride = gridDim.x * blockDim.x;
  for (int i = idx; i < accN; i += stride) acc[i] = 0.f;

  if (idx == 0) {
    float S[5][3][3];
#pragma unroll
    for (int a = 0; a < 5; ++a)
      for (int i = 0; i < 3; ++i)
        for (int j = 0; j < 3; ++j) S[a][i][j] = 0.f;
    // m=-2: xy ; m=-1: yz ; m=0: 2z^2-x^2-y^2 ; m=1: xz ; m=2: x^2-y^2
    S[0][2][0] = S[0][0][2] = R2I;
    S[1][0][1] = S[1][1][0] = R2I;
    S[2][0][0] = -R6I; S[2][1][1] = 2.f*R6I; S[2][2][2] = -R6I;
    S[3][2][1] = S[3][1][2] = R2I;
    S[4][2][2] = R2I;  S[4][0][0] = -R2I;

    float T1[75], T2[125];
    float n1 = 0.f, n2 = 0.f;
    for (int a = 0; a < 5; ++a) {
      for (int b = 0; b < 5; ++b) {
        float M[3][3];
        for (int i = 0; i < 3; ++i)
          for (int j = 0; j < 3; ++j) {
            float v = 0.f;
            for (int k = 0; k < 3; ++k) v += S[a][i][k] * S[b][k][j];
            M[i][j] = v;
          }
        // l=1 part: vector dual of the commutator
        for (int mm = 0; mm < 3; ++mm) {
          float v = 0.f;
          for (int c = 0; c < 3; ++c)
            for (int d = 0; d < 3; ++d) v += epsf(mm, c, d) * M[c][d];
          T1[(a*5 + b)*3 + mm] = v; n1 += v*v;
        }
        // l=2 part: anticommutator projected on S_c
        for (int c = 0; c < 5; ++c) {
          float v = 0.f;
          for (int i = 0; i < 3; ++i)
            for (int j = 0; j < 3; ++j) v += (M[i][j] + M[j][i]) * S[c][i][j];
          T2[(a*5 + b)*5 + c] = v; n2 += v*v;
        }
      }
    }
    float s1 = rsqrtf(n1), s2 = rsqrtf(n2);
    for (int t = 0; t < 75;  ++t) w221[t] = T1[t] * s1;
    for (int t = 0; t < 125; ++t) w222[t] = T2[t] * s2;
  }
}

// ---------------------------------------------------------------------------
// Main kernel: one wave32 per 16-node tile; 2 waves per 64-thread block.
// ---------------------------------------------------------------------------
__global__ __launch_bounds__(64) void tp_main(
    const int* __restrict__ batch,
    const float* __restrict__ xsc,   // (N,128)
    const float* __restrict__ xsp,   // (N,480)
    const float* __restrict__ WU0, const float* __restrict__ WU1,
    const float* __restrict__ WU2, const float* __restrict__ WV0,
    const float* __restrict__ WV1, const float* __restrict__ WV2,
    const float* __restrict__ Wg1, const float* __restrict__ bg1,
    const float* __restrict__ Wg2, const float* __restrict__ bg2,
    const float* __restrict__ Wp0, const float* __restrict__ Wp1,
    const float* __restrict__ Wp2,
    const float* __restrict__ w221, const float* __restrict__ w222,
    float* __restrict__ acc, int N, int numTiles)
{
  __shared__ f16   sH1[2][16 * 64];     // silu(X@Wg1+b1), f16, row-major
  __shared__ float sWK[2][9 * 272];     // [k][node][u], padded stride 17
  __shared__ float sHU[2][9 * 272];     // [comp][node][u]
  __shared__ float sHV[2][9 * 272];

  const int wave = threadIdx.x >> 5;
  const int lane = threadIdx.x & 31;
  const int tile = blockIdx.x * 2 + wave;
  if (tile >= numTiles) return;               // wave-uniform: EXEC stays full

  const int n0   = tile * 16;
  const int m    = lane & 15;                 // A row / D column index
  const int half = lane >> 4;
  const int col  = m;
  int nodeIdx = n0 + m; if (nodeIdx >= N) nodeIdx = N - 1;

  f16*   H1w = sH1[wave];
  float* WKw = sWK[wave];
  float* HUw = sHU[wave];
  float* HVw = sHV[wave];

  // Prefetch next tile's scalar features into cache (CDNA5 global_prefetch).
  if (n0 + 16 < N) __builtin_prefetch(xsc + (size_t)(n0 + 16 + m) * 128, 0, 0);

  // ---- Gate MLP layer 1: H1 = silu(X(16x128) @ Wg1(128x64) + b1) ----------
  v8f acc4[4] = {zero8(), zero8(), zero8(), zero8()};
#pragma unroll
  for (int c = 0; c < 4; ++c) {
    v16h a = load_a_strided(xsc + (size_t)nodeIdx * 128 + c * 32, half, 1);
#pragma unroll
    for (int nb = 0; nb < 4; ++nb) {
      v16h b = load_brow(Wg1 + (size_t)(c * 32 + lane) * 64 + nb * 16, 64);
      acc4[nb] = wmma_f16(a, b, acc4[nb]);
    }
  }
#pragma unroll
  for (int nb = 0; nb < 4; ++nb) {
    float bias = bg1[nb * 16 + col];
#pragma unroll
    for (int r = 0; r < 8; ++r) {
      float x = acc4[nb][r] + bias;
      float h = x / (1.f + __expf(-x));            // silu
      H1w[(8 * half + r) * 64 + nb * 16 + col] = (f16)h;
    }
  }
  asm volatile("s_wait_dscnt 0x0" ::: "memory");   // same-wave LDS RAW

  // ---- Gate MLP layer 2: WK = H1(16x64) @ Wg2(64x144) + b2 ----------------
  {
    v16h ha0 = load_a_lds(H1w + m * 64, half, 0);
    v16h ha1 = load_a_lds(H1w + m * 64, half, 32);
#pragma unroll
    for (int nb = 0; nb < 9; ++nb) {
      v8f aw = zero8();
      aw = wmma_f16(ha0, load_brow(Wg2 + (size_t)lane * 144 + nb * 16, 144), aw);
      aw = wmma_f16(ha1, load_brow(Wg2 + (size_t)(32 + lane) * 144 + nb * 16, 144), aw);
      float bias = bg2[nb * 16 + col];
#pragma unroll
      for (int r = 0; r < 8; ++r)
        WKw[nb * 272 + (8 * half + r) * 17 + col] = aw[r] + bias;
    }
  }

  // ---- Irrep linear layers: hU/hV = xs_l @ W{U,V}_l / sqrt(mul) -----------
  // l = 0 (mul 128, 1 comp, K-chunks 4)
  {
    v8f aU = zero8(), aV = zero8();
#pragma unroll
    for (int c = 0; c < 4; ++c) {
      v16h a = load_a_strided(xsp + (size_t)nodeIdx * 480 + c * 32, half, 1);
      aU = wmma_f16(a, load_brow(WU0 + (size_t)(c * 32 + lane) * 16, 16), aU);
      aV = wmma_f16(a, load_brow(WV0 + (size_t)(c * 32 + lane) * 16, 16), aV);
    }
#pragma unroll
    for (int r = 0; r < 8; ++r) {
      HUw[(8 * half + r) * 17 + col] = aU[r] * 0.08838834765f;  // 1/sqrt(128)
      HVw[(8 * half + r) * 17 + col] = aV[r] * 0.08838834765f;
    }
  }
  // l = 1 (mul 64, 3 comps, stride 3, K-chunks 2)
  {
    v16h bu0 = load_brow(WU1 + (size_t)lane * 16, 16);
    v16h bu1 = load_brow(WU1 + (size_t)(32 + lane) * 16, 16);
    v16h bv0 = load_brow(WV1 + (size_t)lane * 16, 16);
    v16h bv1 = load_brow(WV1 + (size_t)(32 + lane) * 16, 16);
#pragma unroll
    for (int i = 0; i < 3; ++i) {
      const float* base = xsp + (size_t)nodeIdx * 480 + 128 + i;
      v16h a0 = load_a_strided(base,          half, 3);
      v16h a1 = load_a_strided(base + 32 * 3, half, 3);
      v8f aU = zero8(), aV = zero8();
      aU = wmma_f16(a0, bu0, aU); aU = wmma_f16(a1, bu1, aU);
      aV = wmma_f16(a0, bv0, aV); aV = wmma_f16(a1, bv1, aV);
#pragma unroll
      for (int r = 0; r < 8; ++r) {
        HUw[(1 + i) * 272 + (8 * half + r) * 17 + col] = aU[r] * 0.125f;   // 1/sqrt(64)
        HVw[(1 + i) * 272 + (8 * half + r) * 17 + col] = aV[r] * 0.125f;
      }
    }
  }
  // l = 2 (mul 32, 5 comps, stride 5, K-chunks 1)
  {
    v16h bu = load_brow(WU2 + (size_t)lane * 16, 16);
    v16h bv = load_brow(WV2 + (size_t)lane * 16, 16);
#pragma unroll
    for (int i = 0; i < 5; ++i) {
      v16h a = load_a_strided(xsp + (size_t)nodeIdx * 480 + 320 + i, half, 5);
      v8f aU = zero8(), aV = zero8();
      aU = wmma_f16(a, bu, aU);
      aV = wmma_f16(a, bv, aV);
#pragma unroll
      for (int r = 0; r < 8; ++r) {
        HUw[(4 + i) * 272 + (8 * half + r) * 17 + col] = aU[r] * 0.1767766953f; // 1/sqrt(32)
        HVw[(4 + i) * 272 + (8 * half + r) * 17 + col] = aV[r] * 0.1767766953f;
      }
    }
  }
  asm volatile("s_wait_dscnt 0x0" ::: "memory");

  // ---- Tensor-product contraction ----------------------------------------
  // Per-lane: node = lane%16, u-half = lane/16 (8 u's each); combine halves
  // with one shfl_xor(16). All normalizations + Wp folded into c0..c8.
  const int tn = m;
  const int ub = half * 8;
  const int rowoff = tn * 17 + ub;

  const float c0 = 0.14433756730f * Wp0[0];           // (0,0,0)
  const float c1 = 0.125f         * Wp2[0];           // (0,2,2)
  const float c2 = 0.08333333333f * Wp0[1];           // (1,1,0)
  const float c3 = 0.125f         * Wp1[0];           // (1,1,1)
  const float c4 = 0.125f         * Wp2[1];           // (1,1,2)
  const float c5 = 0.125f         * Wp2[2];           // (2,0,2)
  const float c6 = 0.06454972244f * Wp0[2];           // (2,2,0)
  const float c7 = 0.30618621785f * Wp1[1];           // (2,2,1)
  const float c8 = 0.27950849719f * Wp2[3];           // (2,2,2)

  float s[9];
#pragma unroll
  for (int c = 0; c < 9; ++c) s[c] = 0.f;

  float hu0[8], hv0[8];
#pragma unroll
  for (int u = 0; u < 8; ++u) { hu0[u] = HUw[rowoff + u]; hv0[u] = HVw[rowoff + u]; }

  // k0: (0,0,0)
  { float wk[8], p = 0.f;
#pragma unroll
    for (int u = 0; u < 8; ++u) wk[u] = WKw[0 * 272 + rowoff + u];
#pragma unroll
    for (int u = 0; u < 8; ++u) p += wk[u] * hu0[u] * hv0[u];
    s[0] += c0 * p; }

  float hu2[5][8], hv2[5][8];
#pragma unroll
  for (int a = 0; a < 5; ++a)
#pragma unroll
    for (int u = 0; u < 8; ++u) {
      hu2[a][u] = HUw[(4 + a) * 272 + rowoff + u];
      hv2[a][u] = HVw[(4 + a) * 272 + rowoff + u];
    }

  // k1: (0,2,2)  (delta/sqrt5 folded)
  { float wk[8];
#pragma unroll
    for (int u = 0; u < 8; ++u) wk[u] = WKw[1 * 272 + rowoff + u];
#pragma unroll
    for (int mm = 0; mm < 5; ++mm) { float p = 0.f;
#pragma unroll
      for (int u = 0; u < 8; ++u) p += wk[u] * hu0[u] * hv2[mm][u];
      s[4 + mm] += c1 * p; } }

  // k5: (2,0,2)
  { float wk[8];
#pragma unroll
    for (int u = 0; u < 8; ++u) wk[u] = WKw[5 * 272 + rowoff + u];
#pragma unroll
    for (int mm = 0; mm < 5; ++mm) { float p = 0.f;
#pragma unroll
      for (int u = 0; u < 8; ++u) p += wk[u] * hu2[mm][u] * hv0[u];
      s[4 + mm] += c5 * p; } }

  float hu1[3][8], hv1[3][8];
#pragma unroll
  for (int a = 0; a < 3; ++a)
#pragma unroll
    for (int u = 0; u < 8; ++u) {
      hu1[a][u] = HUw[(1 + a) * 272 + rowoff + u];
      hv1[a][u] = HVw[(1 + a) * 272 + rowoff + u];
    }

  // k2: (1,1,0)  (delta/sqrt3 folded)
  { float wk[8], p = 0.f;
#pragma unroll
    for (int u = 0; u < 8; ++u) wk[u] = WKw[2 * 272 + rowoff + u];
#pragma unroll
    for (int u = 0; u < 8; ++u)
      p += wk[u] * (hu1[0][u]*hv1[0][u] + hu1[1][u]*hv1[1][u] + hu1[2][u]*hv1[2][u]);
    s[0] += c2 * p; }

  // k3: (1,1,1)  (epsilon/sqrt6 folded) — cross product in (y,z,x) basis
  { float wk[8], p0 = 0.f, p1 = 0.f, p2 = 0.f;
#pragma unroll
    for (int u = 0; u < 8; ++u) wk[u] = WKw[3 * 272 + rowoff + u];
#pragma unroll
    for (int u = 0; u < 8; ++u) {
      p0 += wk[u] * (hu1[1][u]*hv1[2][u] - hu1[2][u]*hv1[1][u]);
      p1 += wk[u] * (hu1[2][u]*hv1[0][u] - hu1[0][u]*hv1[2][u]);
      p2 += wk[u] * (hu1[0][u]*hv1[1][u] - hu1[1][u]*hv1[0][u]);
    }
    s[1] += c3 * p0; s[2] += c3 * p1; s[3] += c3 * p2; }

  // k4: (1,1,2)  S_m bilinear forms (1/sqrt5 folded into c4)
  { float wk[8], t0=0.f,t1=0.f,t2=0.f,t3=0.f,t4=0.f;
#pragma unroll
    for (int u = 0; u < 8; ++u) wk[u] = WKw[4 * 272 + rowoff + u];
#pragma unroll
    for (int u = 0; u < 8; ++u) {
      float uy = hu1[0][u], uz = hu1[1][u], ux = hu1[2][u];
      float vy = hv1[0][u], vz = hv1[1][u], vx = hv1[2][u];
      float w = wk[u];
      t0 += w * (ux*vy + uy*vx);
      t1 += w * (uy*vz + uz*vy);
      t2 += w * (2.f*uz*vz - uy*vy - ux*vx);
      t3 += w * (ux*vz + uz*vx);
      t4 += w * (ux*vx - uy*vy);
    }
    s[4] += c4 * R2I * t0; s[5] += c4 * R2I * t1; s[6] += c4 * R6I * t2;
    s[7] += c4 * R2I * t3; s[8] += c4 * R2I * t4; }

  // k6: (2,2,0)  (delta/sqrt5 folded)
  { float wk[8], p = 0.f;
#pragma unroll
    for (int u = 0; u < 8; ++u) wk[u] = WKw[6 * 272 + rowoff + u];
#pragma unroll
    for (int u = 0; u < 8; ++u) {
      float d = 0.f;
#pragma unroll
      for (int a = 0; a < 5; ++a) d += hu2[a][u] * hv2[a][u];
      p += wk[u] * d;
    }
    s[0] += c6 * p; }

  // k7: (2,2,1)  runtime table
  { float wk[8];
#pragma unroll
    for (int u = 0; u < 8; ++u) wk[u] = WKw[7 * 272 + rowoff + u];
#pragma unroll
    for (int a = 0; a < 5; ++a)
#pragma unroll
      for (int b = 0; b < 5; ++b) {
        float p = 0.f;
#pragma unroll
        for (int u = 0; u < 8; ++u) p += wk[u] * hu2[a][u] * hv2[b][u];
        p *= c7;
#pragma unroll
        for (int mm = 0; mm < 3; ++mm) s[1 + mm] += w221[(a*5 + b)*3 + mm] * p;
      } }

  // k8: (2,2,2)  runtime table
  { float wk[8];
#pragma unroll
    for (int u = 0; u < 8; ++u) wk[u] = WKw[8 * 272 + rowoff + u];
#pragma unroll
    for (int a = 0; a < 5; ++a)
#pragma unroll
      for (int b = 0; b < 5; ++b) {
        float p = 0.f;
#pragma unroll
        for (int u = 0; u < 8; ++u) p += wk[u] * hu2[a][u] * hv2[b][u];
        p *= c8;
#pragma unroll
        for (int mm = 0; mm < 5; ++mm) s[4 + mm] += w222[(a*5 + b)*5 + mm] * p;
      } }

  // Combine the two u-halves (lane ^ 16 holds the other 8 u's of same node).
#pragma unroll
  for (int c = 0; c < 9; ++c) s[c] += __shfl_xor(s[c], 16, 32);

  // Segment sum: one lane per node does 9 f32 global atomics.
  if (half == 0 && (n0 + tn) < N) {
    int g = batch[n0 + tn];
#pragma unroll
    for (int c = 0; c < 9; ++c) atomicAdd(&acc[(size_t)g * 9 + c], s[c]);
  }
}

// ---------------------------------------------------------------------------
// Finalize: res_cart[g][a][b] = sum_k acc[g][k] * Q_COB[k][CART[a]][CART[b]]
// Q_COB[k=0] = delta/sqrt3 ; k=1..3 = eps/sqrt2 ; k=4..8 = S_m
// ---------------------------------------------------------------------------
__global__ void finalize_kernel(const float* __restrict__ acc,
                                float* __restrict__ out, int G) {
  int t = blockIdx.x * blockDim.x + threadIdx.x;
  if (t >= G * 9) return;
  int g = t / 9, ab = t % 9, a = ab / 3, b = ab % 3;
  const int CART[3] = {2, 0, 1};
  int i = CART[a], j = CART[b];
  const float* r = acc + (size_t)g * 9;

  float v = (i == j) ? r[0] * 0.57735026919f : 0.f;
  v += R2I * (epsf(i, j, 0) * r[1] + epsf(i, j, 1) * r[2] + epsf(i, j, 2) * r[3]);
  // l=2: S_m[i][j] entries
  if ((i == 2 && j == 0) || (i == 0 && j == 2)) v += r[4] * R2I;
  if ((i == 0 && j == 1) || (i == 1 && j == 0)) v += r[5] * R2I;
  if (i == j) v += r[6] * ((i == 1) ? 2.f * R6I : -R6I);
  if ((i == 2 && j == 1) || (i == 1 && j == 2)) v += r[7] * R2I;
  if (i == j && i != 1) v += r[8] * ((i == 2) ? R2I : -R2I);
  out[t] = v;
}

// ---------------------------------------------------------------------------
extern "C" void kernel_launch(void* const* d_in, const int* in_sizes, int n_in,
                              void* d_out, int out_size, void* d_ws, size_t ws_size,
                              hipStream_t stream) {
  const int*   batch = (const int*)  d_in[0];
  const float* xsc   = (const float*)d_in[1];
  const float* xsp   = (const float*)d_in[2];
  const float* WU0   = (const float*)d_in[3];
  const float* WU1   = (const float*)d_in[4];
  const float* WU2   = (const float*)d_in[5];
  const float* WV0   = (const float*)d_in[6];
  const float* WV1   = (const float*)d_in[7];
  const float* WV2   = (const float*)d_in[8];
  const float* Wg1   = (const float*)d_in[9];
  const float* bg1   = (const float*)d_in[10];
  const float* Wg2   = (const float*)d_in[11];
  const float* bg2   = (const float*)d_in[12];
  const float* Wp0   = (const float*)d_in[13];
  const float* Wp1   = (const float*)d_in[14];
  const float* Wp2   = (const float*)d_in[15];

  const int N = in_sizes[0];
  const int G = out_size / 9;

  float* acc  = (float*)d_ws;          // G*9 segment-sum accumulator
  float* w221 = acc + (size_t)G * 9;   // 75 floats
  float* w222 = w221 + 75;             // 125 floats

  setup_kernel<<<64, 256, 0, stream>>>(acc, G * 9, w221, w222);

  const int numTiles = (N + 15) / 16;
  const int blocks   = (numTiles + 1) / 2;   // 2 waves (tiles) per block
  tp_main<<<blocks, 64, 0, stream>>>(batch, xsc, xsp, WU0, WU1, WU2,
                                     WV0, WV1, WV2, Wg1, bg1, Wg2, bg2,
                                     Wp0, Wp1, Wp2, w221, w222, acc, N, numTiles);

  finalize_kernel<<<(G * 9 + 255) / 256, 256, 0, stream>>>(acc, (float*)d_out, G);
}
